// MLA_78391743087078
// MI455X (gfx1250) — matrix operand order repaired and checked by reading, not compiled, not run on hardware
//
#include <hip/hip_runtime.h>
#include <hip/hip_bf16.h>

// ---------------- problem constants ----------------
#define SS        2048
#define BB        2
#define TOK       (BB*SS)      // 4096 tokens
#define DIMX      2048
#define NHEADS    16
#define QK_NOPE   128
#define QK_ROPE   64
#define QKD       192
#define VDIM      128
#define W1N       8192         // 3072 (q) + 3072 (k) + 2048 (v)
#define QNOPE0    0
#define QROPE0    2048
#define KNOPE0    3072
#define KROPE0    5120
#define VCOL0     6144
#define OUTW      2048         // NHEADS*VDIM

typedef __bf16 bf16_t;
typedef __attribute__((ext_vector_type(16))) __bf16 v16bf;
typedef __attribute__((ext_vector_type(8)))  float  v8f;

union ABfrag { v16bf v; unsigned int u[8]; };

__device__ __forceinline__ unsigned short f2bf(float f) {
  unsigned int u = __float_as_uint(f);
  u += 0x7FFFu + ((u >> 16) & 1u);           // round-to-nearest-even
  return (unsigned short)(u >> 16);
}
__device__ __forceinline__ float bf2f(unsigned short h) {
  return __uint_as_float(((unsigned int)h) << 16);
}
__device__ __forceinline__ unsigned int ld_u32(const unsigned short* p) {
  return *(const unsigned int*)p;
}

// CDNA5 async DMA: global memory -> LDS, 16B per lane, tracked by ASYNCcnt.
// GV addressing mode: vdst = LDS byte address, vaddr = 64-bit global address.
__device__ __forceinline__ void async_copy_b128(unsigned lds_addr,
                                                const void* gaddr) {
  asm volatile("global_load_async_to_lds_b128 %0, %1, off"
               :: "v"(lds_addr), "v"((unsigned long long)(size_t)gaddr)
               : "memory");
}
__device__ __forceinline__ void wait_async0() {
  asm volatile("s_wait_asynccnt 0x0" ::: "memory");
}
__device__ __forceinline__ unsigned lds_addr_of(const void* p) {
  return (unsigned)(size_t)p;    // addrspace(3) ptr -> LDS byte offset
}

// ---------------- elementwise conversion kernels ----------------
__global__ void cvt_f32_bf16_kernel(const float* __restrict__ src,
                                    unsigned short* __restrict__ dst, int n) {
  int i = blockIdx.x * blockDim.x + threadIdx.x;
  if (i < n) dst[i] = f2bf(src[i]);
}

// pack wq|wk|wv into one [DIMX][W1N] bf16 matrix
__global__ void pack_w1_kernel(const float* __restrict__ wq,
                               const float* __restrict__ wk,
                               const float* __restrict__ wv,
                               unsigned short* __restrict__ w1) {
  int i = blockIdx.x * blockDim.x + threadIdx.x;
  if (i >= DIMX * W1N) return;
  int row = i >> 13;          // /8192
  int col = i & (W1N - 1);
  float v;
  if (col < 3072)      v = wq[row * 3072 + col];
  else if (col < 6144) v = wk[row * 3072 + (col - 3072)];
  else                 v = wv[row * 2048 + (col - 6144)];
  w1[i] = f2bf(v);
}

// ---------------- RoPE (in place on bf16 qkv buffer) ----------------
__global__ void rope_kernel(unsigned short* __restrict__ qkv,
                            const float* __restrict__ freqs) {
  const int total = TOK * NHEADS * (QK_ROPE / 2);   // pairs per tensor
  int idx = blockIdx.x * blockDim.x + threadIdx.x;
  if (idx >= 2 * total) return;
  int which = idx / total;                // 0 = q-rope, 1 = k-rope
  int rem   = idx % total;
  int j   = rem & 31;                     // pair index within head
  int hh  = (rem >> 5) & (NHEADS - 1);
  int tok = rem >> 9;                     // b*S + s
  int s   = tok & (SS - 1);
  float ang = freqs[s * 32 + j];
  float sn, cs;
  __sincosf(ang, &sn, &cs);
  int colbase = (which == 0 ? QROPE0 : KROPE0) + hh * QK_ROPE + 2 * j;
  unsigned short* p = qkv + (size_t)tok * W1N + colbase;
  float x0 = bf2f(p[0]), x1 = bf2f(p[1]);
  p[0] = f2bf(x0 * cs - x1 * sn);
  p[1] = f2bf(x0 * sn + x1 * cs);
}

// ---------------- tiled bf16 GEMM: C[M,N] = A[M,K] * B[K,N] ----------------
// block = 256 threads (8 waves), tile 128x128, K-step 32.
// Tile staging uses CDNA5 global_load_async_to_lds_b128 (ASYNCcnt) plus
// global_prefetch for the next K-slab.
template<bool OUT_BF16>
__global__ __launch_bounds__(256)
void gemm_bf16_kernel(const unsigned short* __restrict__ A,
                      const unsigned short* __restrict__ B,
                      void* __restrict__ Cptr,
                      int M, int N, int K) {
  constexpr int BM = 128, BN = 128, BK = 32;
  __shared__ unsigned short As[BM][BK + 8];   // row stride 80B (16B-aligned)
  __shared__ unsigned short Bs[BK][BN + 8];   // row stride 272B (16B-aligned)

  const int tid  = threadIdx.x;
  const int wave = tid >> 5;
  const int lane = tid & 31;
  const int hlf  = lane >> 4;
  const int l16  = lane & 15;

  const int bm = blockIdx.y * BM;
  const int bn = blockIdx.x * BN;
  const int wm = (wave & 3) * 32;   // wave M offset in tile
  const int wn = (wave >> 2) * 64;  // wave N offset in tile

  // per-thread staging coordinates
  const int arow = tid >> 1, acol = (tid & 1) * 16;
  const int brow = tid >> 3, bcol = (tid & 7) * 16;
  const unsigned ldsA = lds_addr_of(&As[arow][acol]);
  const unsigned ldsB = lds_addr_of(&Bs[brow][bcol]);

  v8f acc[2][4];
  for (int i = 0; i < 2; ++i)
    for (int j = 0; j < 4; ++j)
      for (int e = 0; e < 8; ++e) acc[i][j][e] = 0.f;

  for (int k0 = 0; k0 < K; k0 += BK) {
    // async DMA of A/B tiles straight into LDS (no VGPR round-trip)
    const unsigned short* srcA = A + (size_t)(bm + arow) * K + k0 + acol;
    const unsigned short* srcB = B + (size_t)(k0 + brow) * N + bn + bcol;
    async_copy_b128(ldsA,      srcA);
    async_copy_b128(ldsA + 16, srcA + 8);
    async_copy_b128(ldsB,      srcB);
    async_copy_b128(ldsB + 16, srcB + 8);
    // prefetch next K-slab into L2 while this tile computes
    if (k0 + BK < K) {
      __builtin_prefetch(srcA + BK, 0, 0);
      __builtin_prefetch(srcB + (size_t)BK * N, 0, 0);
    }
    wait_async0();
    __syncthreads();

    ABfrag afr[2];
    #pragma unroll
    for (int im = 0; im < 2; ++im) {
      int row = wm + im * 16 + l16;
      #pragma unroll
      for (int i = 0; i < 8; ++i) {
        int kk = 2 * i + hlf * 16;   // contiguous k order (matches B)
        afr[im].u[i] = ld_u32(&As[row][kk]);
      }
    }
    #pragma unroll
    for (int in = 0; in < 4; ++in) {
      ABfrag bfr;
      int col = wn + in * 16 + l16;
      int kb  = hlf * 16;
      #pragma unroll
      for (int i = 0; i < 8; ++i) {
        unsigned int lo = Bs[kb + 2 * i][col];
        unsigned int hi = Bs[kb + 2 * i + 1][col];
        bfr.u[i] = lo | (hi << 16);
      }
      #pragma unroll
      for (int im = 0; im < 2; ++im)
        acc[im][in] = __builtin_amdgcn_wmma_f32_16x16x32_bf16(
            false, afr[im].v, false, bfr.v, (short)0, acc[im][in], false, false);
    }
    __syncthreads();
  }

  #pragma unroll
  for (int im = 0; im < 2; ++im)
    #pragma unroll
    for (int in = 0; in < 4; ++in) {
      int gcol = bn + wn + in * 16 + l16;
      #pragma unroll
      for (int v = 0; v < 8; ++v) {
        int grow = bm + wm + im * 16 + v + hlf * 8;   // ISA C/D layout
        float val = acc[im][in][v];
        if (OUT_BF16)
          ((unsigned short*)Cptr)[(size_t)grow * N + gcol] = f2bf(val);
        else
          ((float*)Cptr)[(size_t)grow * N + gcol] = val;
      }
    }
}

// ---------------- flash attention (causal) ----------------
// grid: (S/128, NHEADS, B); block 256 = 8 waves; wave owns 16 q rows.
__global__ __launch_bounds__(256)
void attention_kernel(const unsigned short* __restrict__ qkv,   // [TOK][W1N] bf16
                      unsigned short* __restrict__ attn_out) {  // [TOK][OUTW] bf16
  __shared__ unsigned short Ks[64][QKD + 8];     // [key][d]   row 400B
  __shared__ unsigned short Vst[VDIM][64 + 8];   // [dv][key]  row 144B
  __shared__ unsigned short Ps[8][16][64 + 8];   // per-wave P row 144B

  const int tid  = threadIdx.x;
  const int wave = tid >> 5;
  const int lane = tid & 31;
  const int hlf  = lane >> 4;
  const int l16  = lane & 15;

  const int qt = blockIdx.x;
  const int h  = blockIdx.y;
  const int b  = blockIdx.z;
  const int qbase = qt * 128 + wave * 16;        // first q row of wave (in S)
  const size_t rowbase = (size_t)b * SS;

  // Q fragments: 6 chunks of K=32 covering d=0..191, loaded from global
  ABfrag qf[6];
  {
    int r = qbase + l16;
    const unsigned short* qrowp = qkv + (rowbase + r) * W1N;
    #pragma unroll
    for (int c = 0; c < 6; ++c) {
      #pragma unroll
      for (int i = 0; i < 8; ++i) {
        int kk  = c * 32 + 2 * i + hlf * 16;
        int col = (kk < QK_NOPE) ? (QNOPE0 + h * QK_NOPE + kk)
                                 : (QROPE0 + h * QK_ROPE + (kk - QK_NOPE));
        qf[c].u[i] = ld_u32(qrowp + col);
      }
    }
  }

  float m_run[8], l_run[8];
  v8f oacc[8];
  #pragma unroll
  for (int v = 0; v < 8; ++v) { m_run[v] = -1e30f; l_run[v] = 0.f; }
  #pragma unroll
  for (int vf = 0; vf < 8; ++vf)
    for (int e = 0; e < 8; ++e) oacc[vf][e] = 0.f;

  const float scale = 0.07216878364870322f;      // 1/sqrt(192)
  const int ktiles = qt * 2 + 2;                 // causal bound

  for (int kt = 0; kt < ktiles; ++kt) {
    const int kb0 = kt * 64;
    // K tile [64 keys][192 d]: layout-preserving -> async DMA to LDS
    for (int idx = tid; idx < 64 * 24; idx += 256) {
      int key = idx / 24;
      int d   = (idx % 24) * 8;
      int col = (d < QK_NOPE) ? (KNOPE0 + h * QK_NOPE + d)
                              : (KROPE0 + h * QK_ROPE + (d - QK_NOPE));
      async_copy_b128(lds_addr_of(&Ks[key][d]),
                      qkv + (rowbase + kb0 + key) * W1N + col);
    }
    // V tile transposed into Vst[dv][key] (manual: async can't transpose)
    for (int idx = tid; idx < 64 * 16; idx += 256) {
      int key = idx / 16;
      int d   = (idx % 16) * 8;
      uint4 v4 = *(const uint4*)(qkv + (rowbase + kb0 + key) * W1N +
                                 VCOL0 + h * VDIM + d);
      const unsigned short* e = (const unsigned short*)&v4;
      #pragma unroll
      for (int j = 0; j < 8; ++j) Vst[d + j][key] = e[j];
    }
    wait_async0();
    __syncthreads();

    // scores: S = Q(16x192) * K^T(192x64)  -> 4 accum frags
    v8f sacc[4];
    #pragma unroll
    for (int nf = 0; nf < 4; ++nf)
      for (int e = 0; e < 8; ++e) sacc[nf][e] = 0.f;
    #pragma unroll
    for (int c = 0; c < 6; ++c) {
      #pragma unroll
      for (int nf = 0; nf < 4; ++nf) {
        ABfrag bfr;
        int key = nf * 16 + l16;
        int kb  = c * 32 + hlf * 16;
        #pragma unroll
        for (int i = 0; i < 8; ++i)
          bfr.u[i] = ld_u32(&Ks[key][kb + 2 * i]);   // contiguous d-pair
        sacc[nf] = __builtin_amdgcn_wmma_f32_16x16x32_bf16(
            false, qf[c].v, false, bfr.v, (short)0, sacc[nf], false, false);
      }
    }

    // causal mask + scale + online softmax (per row v, cols across 16 lanes)
    #pragma unroll
    for (int v = 0; v < 8; ++v) {
      const int qrow = qbase + v + hlf * 8;
      float mx = -1e30f;
      #pragma unroll
      for (int nf = 0; nf < 4; ++nf) {
        int key = kb0 + nf * 16 + l16;
        float s = sacc[nf][v] * scale;
        if (key > qrow) s = -1e30f;
        sacc[nf][v] = s;
        mx = fmaxf(mx, s);
      }
      #pragma unroll
      for (int off = 1; off < 16; off <<= 1)
        mx = fmaxf(mx, __shfl_xor(mx, off, 32));
      float mnew  = fmaxf(m_run[v], mx);
      float alpha = __expf(m_run[v] - mnew);
      float rs = 0.f;
      #pragma unroll
      for (int nf = 0; nf < 4; ++nf) {
        float p = __expf(sacc[nf][v] - mnew);
        sacc[nf][v] = p;
        rs += p;
      }
      #pragma unroll
      for (int off = 1; off < 16; off <<= 1)
        rs += __shfl_xor(rs, off, 32);
      l_run[v] = l_run[v] * alpha + rs;
      m_run[v] = mnew;
      #pragma unroll
      for (int vf = 0; vf < 8; ++vf) oacc[vf][v] *= alpha;
    }

    // P (C layout) -> per-wave LDS -> reload as A fragments (bf16)
    #pragma unroll
    for (int nf = 0; nf < 4; ++nf)
      #pragma unroll
      for (int v = 0; v < 8; ++v)
        Ps[wave][v + hlf * 8][nf * 16 + l16] = f2bf(sacc[nf][v]);

    #pragma unroll
    for (int pf = 0; pf < 2; ++pf) {
      ABfrag pfr;
      #pragma unroll
      for (int i = 0; i < 8; ++i) {
        int kk = pf * 32 + 2 * i + hlf * 16;
        pfr.u[i] = ld_u32(&Ps[wave][l16][kk]);
      }
      #pragma unroll
      for (int vf = 0; vf < 8; ++vf) {
        ABfrag bfr;
        int dv = vf * 16 + l16;
        int kb = pf * 32 + hlf * 16;
        #pragma unroll
        for (int i = 0; i < 8; ++i)
          bfr.u[i] = ld_u32(&Vst[dv][kb + 2 * i]);   // contiguous key-pair
        oacc[vf] = __builtin_amdgcn_wmma_f32_16x16x32_bf16(
            false, pfr.v, false, bfr.v, (short)0, oacc[vf], false, false);
      }
    }
    __syncthreads();   // before next tile overwrites Ks/Vst
  }

  // normalize and store bf16 attention output
  #pragma unroll
  for (int vf = 0; vf < 8; ++vf) {
    int col = h * VDIM + vf * 16 + l16;
    #pragma unroll
    for (int v = 0; v < 8; ++v) {
      int qrow = qbase + v + hlf * 8;
      float val = oacc[vf][v] / l_run[v];
      attn_out[(rowbase + qrow) * OUTW + col] = f2bf(val);
    }
  }
}

// ---------------- launcher ----------------
extern "C" void kernel_launch(void* const* d_in, const int* in_sizes, int n_in,
                              void* d_out, int out_size, void* d_ws, size_t ws_size,
                              hipStream_t stream) {
  (void)in_sizes; (void)n_in; (void)out_size; (void)ws_size;
  const float* x     = (const float*)d_in[0];
  // d_in[1] = start_pos (0, unused), d_in[3] = mask (causal handled in-kernel)
  const float* freqs = (const float*)d_in[2];
  const float* wq    = (const float*)d_in[4];
  const float* wk    = (const float*)d_in[5];
  const float* wv    = (const float*)d_in[6];
  const float* wo    = (const float*)d_in[7];
  float* out = (float*)d_out;

  char* ws = (char*)d_ws;
  unsigned short* x_bf  = (unsigned short*)(ws);                        // 16 MiB
  unsigned short* w1_bf = (unsigned short*)(ws + 16777216);             // 32 MiB
  unsigned short* wo_bf = (unsigned short*)(ws + 16777216 + 33554432);  //  8 MiB
  unsigned short* qkv   = (unsigned short*)(ws + 58720256);             // 64 MiB
  unsigned short* attn  = (unsigned short*)(ws + 125829120);            // 16 MiB

  // 1) precision conversion / packing
  cvt_f32_bf16_kernel<<<(TOK * DIMX + 255) / 256, 256, 0, stream>>>(
      x, x_bf, TOK * DIMX);
  pack_w1_kernel<<<(DIMX * W1N + 255) / 256, 256, 0, stream>>>(
      wq, wk, wv, w1_bf);
  cvt_f32_bf16_kernel<<<(OUTW * DIMX + 255) / 256, 256, 0, stream>>>(
      wo, wo_bf, OUTW * DIMX);

  // 2) fused QKV projection: [4096,2048] x [2048,8192] -> bf16
  gemm_bf16_kernel<true><<<dim3(W1N / 128, TOK / 128), 256, 0, stream>>>(
      x_bf, w1_bf, qkv, TOK, W1N, DIMX);

  // 3) RoPE on q-rope / k-rope slices
  rope_kernel<<<(2 * TOK * NHEADS * (QK_ROPE / 2) + 255) / 256, 256, 0, stream>>>(
      qkv, freqs);

  // 4) causal flash attention
  attention_kernel<<<dim3(SS / 128, NHEADS, BB), 256, 0, stream>>>(qkv, attn);

  // 5) output projection: [4096,2048] x [2048,2048] -> fp32 out
  gemm_bf16_kernel<false><<<dim3(DIMX / 128, TOK / 128), 256, 0, stream>>>(
      attn, wo_bf, out, TOK, DIMX, DIMX);
}
